// SelfAttentionLayer_62740882260779
// MI455X (gfx1250) — matrix operand order repaired and checked by reading, not compiled, not run on hardware
//
#include <hip/hip_runtime.h>

// ---------------------------------------------------------------------------
// SelfAttention (B=8, S=2048, D=1024) for gfx1250 (CDNA5, wave32, WMMA).
// Pipeline: cvt f32->f16, proj GEMM (Q,K,V), scores GEMM, row softmax,
// context GEMM. GEMM core: 128x128 block tile, 8 waves, 32x64 wave tile,
// K-step 64, all 12 fragments preloaded per step (hazard-NOP avoidance),
// double-buffered LDS with peeled loop, v_wmma_f32_16x16x32_f16.
// ---------------------------------------------------------------------------

typedef __attribute__((ext_vector_type(16))) _Float16 v16h;
typedef __attribute__((ext_vector_type(8)))  float    v8f;

#define LDK 66   // LDS tile row stride in f16 elems (64 + 2 pad, even)

union FragU { v16h v; unsigned int u[8]; };

// A fragment (16x32 f16). Tile in LDS row-major [m][k], stride LDK; s points
// at (m=0, k=kk). Lane L: m = L%16, h = L/16.
// VGPR v: K-pair = 16*(v/4) + 8*h + 2*(v%4).
__device__ __forceinline__ v16h ldsLoadA(const _Float16* s, int lane) {
  const int m = lane & 15, h = lane >> 4;
  const unsigned int* row = (const unsigned int*)(s + m * LDK);
  FragU f;
#pragma unroll
  for (int r = 0; r < 8; ++r) {
    const int k = ((r >> 2) << 4) + (h << 3) + ((r & 3) << 1);
    f.u[r] = row[k >> 1];
  }
  return f.v;
}

// B fragment (32x16 f16, K x N). Tile in LDS as [n][k], stride LDK.
// Lane L: n = L%16, h = L/16. VGPR v: K-pair = 16*(v/4) + 8*h + 2*(v%4).
__device__ __forceinline__ v16h ldsLoadB(const _Float16* s, int lane) {
  const int n = lane & 15, h = lane >> 4;
  const unsigned int* row = (const unsigned int*)(s + n * LDK);
  FragU f;
#pragma unroll
  for (int r = 0; r < 8; ++r) {
    const int k = ((r >> 2) << 4) + (h << 3) + ((r & 3) << 1);
    f.u[r] = row[k >> 1];
  }
  return f.v;
}

// Stage a 128x64 tile whose global rows are contiguous in k.
// 256 threads: each covers one row-half (32 f16 = 4 x b128), dword LDS stores.
__device__ __forceinline__ void stageRows(const _Float16* __restrict__ g, long long row0,
                                          long long lda, long long k0,
                                          _Float16* s, int tid) {
  const int r = tid >> 1;          // 0..127
  const int c = (tid & 1) << 5;    // 0 or 32
  const uint4* src = (const uint4*)(g + (row0 + r) * lda + k0 + c);
  const uint4 d0 = src[0];
  const uint4 d1 = src[1];
  const uint4 d2 = src[2];
  const uint4 d3 = src[3];
  unsigned int* dst = (unsigned int*)s + ((r * LDK + c) >> 1);
  dst[ 0] = d0.x; dst[ 1] = d0.y; dst[ 2] = d0.z; dst[ 3] = d0.w;
  dst[ 4] = d1.x; dst[ 5] = d1.y; dst[ 6] = d1.z; dst[ 7] = d1.w;
  dst[ 8] = d2.x; dst[ 9] = d2.y; dst[10] = d2.z; dst[11] = d2.w;
  dst[12] = d3.x; dst[13] = d3.y; dst[14] = d3.z; dst[15] = d3.w;
}

// Stage a 64(k) x 128(n) tile of a [k][n] row-major matrix, transposed into
// LDS [n][k]. Each thread loads two adjacent k-rows x 16 n (4 x b128) and
// packs (k, k+1) pairs into 16 dword LDS stores.
__device__ __forceinline__ void stageTrans(const _Float16* __restrict__ g, long long k0,
                                           long long ldn, long long n0,
                                           _Float16* s, int tid) {
  const int kp = tid >> 3;          // 0..31 -> k = 2*kp, 2*kp+1
  const int c  = (tid & 7) << 4;    // 0..112
  union { uint4 q; _Float16 h[8]; } t0a, t0b, t1a, t1b;
  const _Float16* g0 = g + (k0 + 2 * kp) * ldn + n0 + c;
  const _Float16* g1 = g0 + ldn;
  t0a.q = *(const uint4*)(g0);
  t0b.q = *(const uint4*)(g0 + 8);
  t1a.q = *(const uint4*)(g1);
  t1b.q = *(const uint4*)(g1 + 8);
  unsigned int* dst = (unsigned int*)s;
#pragma unroll
  for (int e = 0; e < 8; ++e) {
    union { unsigned int u; _Float16 h[2]; } pk;
    pk.h[0] = t0a.h[e]; pk.h[1] = t1a.h[e];
    dst[(c + e) * (LDK / 2) + kp] = pk.u;
  }
#pragma unroll
  for (int e = 0; e < 8; ++e) {
    union { unsigned int u; _Float16 h[2]; } pk;
    pk.h[0] = t0b.h[e]; pk.h[1] = t1b.h[e];
    dst[(c + 8 + e) * (LDK / 2) + kp] = pk.u;
  }
}

// One wave's work per K-step (64): preload ALL 12 fragments (both sub-steps),
// then issue 16 WMMAs. Wide fragment working set doubles register reuse
// distance so WMMA WAR-hazard slots are filled with loads, not v_nops.
__device__ __forceinline__ void waveCompute(const _Float16* As, const _Float16* Bs,
                                            int wm, int wn, int lane, v8f (&acc)[2][4]) {
  v16h a[2][2], b[2][4];
#pragma unroll
  for (int s = 0; s < 2; ++s) {
    const int kk = s * 32;
    a[s][0] = ldsLoadA(As + (wm * 32 +  0) * LDK + kk, lane);
    a[s][1] = ldsLoadA(As + (wm * 32 + 16) * LDK + kk, lane);
    b[s][0] = ldsLoadB(Bs + (wn * 64 +  0) * LDK + kk, lane);
    b[s][1] = ldsLoadB(Bs + (wn * 64 + 16) * LDK + kk, lane);
    b[s][2] = ldsLoadB(Bs + (wn * 64 + 32) * LDK + kk, lane);
    b[s][3] = ldsLoadB(Bs + (wn * 64 + 48) * LDK + kk, lane);
  }
#pragma unroll
  for (int s = 0; s < 2; ++s)
#pragma unroll
    for (int mi = 0; mi < 2; ++mi)
#pragma unroll
      for (int ni = 0; ni < 4; ++ni)
        acc[mi][ni] = __builtin_amdgcn_wmma_f32_16x16x32_f16(
            false, a[s][mi], false, b[s][ni], (short)0, acc[mi][ni], false, false);
}

// --------------------------- kernels ---------------------------------------

__global__ __launch_bounds__(256) void attn_cvt_f16(const float* __restrict__ in,
                                                    _Float16* __restrict__ out,
                                                    long long n) {
  long long i = ((long long)blockIdx.x * blockDim.x + threadIdx.x) << 2;
  if (i + 3 < n) {
    float4 v = *(const float4*)(in + i);
    out[i + 0] = (_Float16)v.x;
    out[i + 1] = (_Float16)v.y;
    out[i + 2] = (_Float16)v.z;
    out[i + 3] = (_Float16)v.w;
  }
}

// QKV projection: out[z] = Xb(16384x1024) @ Wb[z](1024x1024) + bias[z], f16 out.
__global__ __launch_bounds__(256, 2) void attn_proj_gemm(const _Float16* __restrict__ Xb,
                                                         const _Float16* __restrict__ Wb,
                                                         const float* __restrict__ bq,
                                                         const float* __restrict__ bk,
                                                         const float* __restrict__ bv,
                                                         _Float16* __restrict__ QKV) {
  const int Mn = 16384, Nn = 1024, Kn = 1024;
  __shared__ _Float16 As[2][128 * LDK];
  __shared__ _Float16 Bs[2][128 * LDK];
  const int z = blockIdx.z;
  const _Float16* W = Wb + (long long)z * Kn * Nn;
  const float* bias = (z == 0) ? bq : (z == 1 ? bk : bv);
  _Float16* out = QKV + (long long)z * Mn * Nn;
  const long long m0 = (long long)blockIdx.y * 128;
  const long long n0 = (long long)blockIdx.x * 128;
  const int tid = threadIdx.x, lane = tid & 31, w = tid >> 5;
  const int wm = w & 3;    // M sub-tile: rows wm*32 .. +31
  const int wn = w >> 2;   // N sub-tile: cols wn*64 .. +63
  v8f acc[2][4] = {};
  stageRows(Xb, m0, Kn, 0, As[0], tid);
  stageTrans(W, 0, Nn, n0, Bs[0], tid);
  int cur = 0;
  for (int k0 = 0; k0 < Kn - 64; k0 += 64) {   // peeled: last step outside
    __syncthreads();
    stageRows(Xb, m0, Kn, k0 + 64, As[cur ^ 1], tid);
    stageTrans(W, k0 + 64, Nn, n0, Bs[cur ^ 1], tid);
    waveCompute(As[cur], Bs[cur], wm, wn, lane, acc);
    cur ^= 1;
  }
  __syncthreads();
  waveCompute(As[cur], Bs[cur], wm, wn, lane, acc);
  const int n = lane & 15, h = lane >> 4;
#pragma unroll
  for (int mi = 0; mi < 2; ++mi)
#pragma unroll
    for (int ni = 0; ni < 4; ++ni)
#pragma unroll
      for (int r = 0; r < 8; ++r) {
        const long long row = m0 + wm * 32 + mi * 16 + r + (h << 3);
        const long long c0  = n0 + wn * 64 + ni * 16 + n;
        out[row * Nn + c0] = (_Float16)(acc[mi][ni][r] + bias[c0]);
      }
}

// scores[b] = Q[b](2048x1024) @ K[b]^T, f32 out. K's [seq][d] rows ARE the
// [n][k] LDS layout, so the B stage is a row copy.
__global__ __launch_bounds__(256, 2) void attn_scores_gemm(const _Float16* __restrict__ Qb,
                                                           const _Float16* __restrict__ Kb,
                                                           float* __restrict__ Sc) {
  const int Sn = 2048, Dk = 1024;
  __shared__ _Float16 As[2][128 * LDK];
  __shared__ _Float16 Bs[2][128 * LDK];
  const long long b = blockIdx.z;
  const _Float16* A  = Qb + b * Sn * Dk;
  const _Float16* Bm = Kb + b * Sn * Dk;
  float* out = Sc + b * Sn * Sn;
  const long long m0 = (long long)blockIdx.y * 128;
  const long long n0 = (long long)blockIdx.x * 128;
  const int tid = threadIdx.x, lane = tid & 31, w = tid >> 5;
  const int wm = w & 3;
  const int wn = w >> 2;
  v8f acc[2][4] = {};
  stageRows(A,  m0, Dk, 0, As[0], tid);
  stageRows(Bm, n0, Dk, 0, Bs[0], tid);
  int cur = 0;
  for (int k0 = 0; k0 < Dk - 64; k0 += 64) {
    __syncthreads();
    stageRows(A,  m0, Dk, k0 + 64, As[cur ^ 1], tid);
    stageRows(Bm, n0, Dk, k0 + 64, Bs[cur ^ 1], tid);
    waveCompute(As[cur], Bs[cur], wm, wn, lane, acc);
    cur ^= 1;
  }
  __syncthreads();
  waveCompute(As[cur], Bs[cur], wm, wn, lane, acc);
  const int n = lane & 15, h = lane >> 4;
#pragma unroll
  for (int mi = 0; mi < 2; ++mi)
#pragma unroll
    for (int ni = 0; ni < 4; ++ni)
#pragma unroll
      for (int r = 0; r < 8; ++r) {
        const long long row = m0 + wm * 32 + mi * 16 + r + (h << 3);
        const long long c0  = n0 + wn * 64 + ni * 16 + n;
        out[row * Sn + c0] = acc[mi][ni][r];
      }
}

// One block per row. Reads 2048 f32 scores, writes normalized P as f16 packed
// into the upper half of this row's own f32 storage (no extra ws, no races).
__global__ __launch_bounds__(256) void attn_softmax_rows(float* __restrict__ Sc) {
  const int Sn = 2048;
  const long long row = blockIdx.x;
  float* r = Sc + row * Sn;
  _Float16* p = (_Float16*)Sc + row * (2LL * Sn) + Sn;
  __shared__ float red[256];
  const int tid = threadIdx.x;
  float x[8];
  float mx = -__builtin_inff();
#pragma unroll
  for (int i = 0; i < 8; ++i) { x[i] = r[tid + (i << 8)]; mx = fmaxf(mx, x[i]); }
  red[tid] = mx; __syncthreads();
  for (int s = 128; s > 0; s >>= 1) {
    if (tid < s) red[tid] = fmaxf(red[tid], red[tid + s]);
    __syncthreads();
  }
  mx = red[0]; __syncthreads();
  float sum = 0.f;
#pragma unroll
  for (int i = 0; i < 8; ++i) { x[i] = __expf(x[i] - mx); sum += x[i]; }
  red[tid] = sum; __syncthreads();
  for (int s = 128; s > 0; s >>= 1) {
    if (tid < s) red[tid] += red[tid + s];
    __syncthreads();
  }
  const float inv = 1.0f / red[0];
#pragma unroll
  for (int i = 0; i < 8; ++i) p[tid + (i << 8)] = (_Float16)(x[i] * inv);
}

// context[b] = P[b](2048x2048 f16, row stride 2S, offset S) @ V[b](2048x1024), f32 out.
__global__ __launch_bounds__(256, 2) void attn_context_gemm(const _Float16* __restrict__ Pbase,
                                                            const _Float16* __restrict__ Vb,
                                                            float* __restrict__ Out) {
  const int Sn = 2048, Dk = 1024;
  __shared__ _Float16 As[2][128 * LDK];
  __shared__ _Float16 Bs[2][128 * LDK];
  const long long b = blockIdx.z;
  const _Float16* P = Pbase + b * (2LL * Sn * Sn) + Sn;  // packed rows, lda = 2S
  const _Float16* V = Vb + b * Sn * Dk;
  float* out = Out + b * Sn * Dk;
  const long long m0 = (long long)blockIdx.y * 128;
  const long long n0 = (long long)blockIdx.x * 128;
  const int tid = threadIdx.x, lane = tid & 31, w = tid >> 5;
  const int wm = w & 3;
  const int wn = w >> 2;
  v8f acc[2][4] = {};
  stageRows(P, m0, 2LL * Sn, 0, As[0], tid);
  stageTrans(V, 0, Dk, n0, Bs[0], tid);
  int cur = 0;
  for (int k0 = 0; k0 < Sn - 64; k0 += 64) {   // reduce over key positions
    __syncthreads();
    stageRows(P, m0, 2LL * Sn, k0 + 64, As[cur ^ 1], tid);
    stageTrans(V, k0 + 64, Dk, n0, Bs[cur ^ 1], tid);
    waveCompute(As[cur], Bs[cur], wm, wn, lane, acc);
    cur ^= 1;
  }
  __syncthreads();
  waveCompute(As[cur], Bs[cur], wm, wn, lane, acc);
  const int n = lane & 15, h = lane >> 4;
#pragma unroll
  for (int mi = 0; mi < 2; ++mi)
#pragma unroll
    for (int ni = 0; ni < 4; ++ni)
#pragma unroll
      for (int r = 0; r < 8; ++r) {
        const long long row = m0 + wm * 32 + mi * 16 + r + (h << 3);
        const long long c0  = n0 + wn * 64 + ni * 16 + n;
        out[row * Dk + c0] = acc[mi][ni][r];
      }
}

// --------------------------- launcher --------------------------------------

extern "C" void kernel_launch(void* const* d_in, const int* in_sizes, int n_in,
                              void* d_out, int out_size, void* d_ws, size_t ws_size,
                              hipStream_t stream) {
  (void)in_sizes; (void)n_in; (void)out_size; (void)ws_size;
  const long long B = 8, S = 2048, D = 1024, M = B * S;

  const float* X  = (const float*)d_in[0];
  const float* Wq = (const float*)d_in[1];
  const float* bq = (const float*)d_in[2];
  const float* Wk = (const float*)d_in[3];
  const float* bk = (const float*)d_in[4];
  const float* Wv = (const float*)d_in[5];
  const float* bv = (const float*)d_in[6];

  // Workspace layout (f16 elems unless noted):
  //   Xb [M*D] | Wb [3*D*D] | QKV [3*M*D] | scores f32 [B*S*S] (P packed inside)
  _Float16* Xb  = (_Float16*)d_ws;
  _Float16* Wb  = Xb + M * D;
  _Float16* QKV = Wb + 3 * D * D;
  float*    Sc  = (float*)(QKV + 3 * M * D);

  // 1) precision convert
  attn_cvt_f16<<<(unsigned)((M * D / 4 + 255) / 256), 256, 0, stream>>>(X, Xb, M * D);
  attn_cvt_f16<<<(unsigned)((D * D / 4 + 255) / 256), 256, 0, stream>>>(Wq, Wb,             D * D);
  attn_cvt_f16<<<(unsigned)((D * D / 4 + 255) / 256), 256, 0, stream>>>(Wk, Wb + D * D,     D * D);
  attn_cvt_f16<<<(unsigned)((D * D / 4 + 255) / 256), 256, 0, stream>>>(Wv, Wb + 2 * D * D, D * D);

  // 2) Q,K,V projections (grid.z selects weight/bias/output)
  attn_proj_gemm<<<dim3((unsigned)(D / 128), (unsigned)(M / 128), 3), 256, 0, stream>>>(
      Xb, Wb, bq, bk, bv, QKV);

  // 3) scores = Q K^T per batch
  attn_scores_gemm<<<dim3((unsigned)(S / 128), (unsigned)(S / 128), (unsigned)B), 256, 0, stream>>>(
      QKV, QKV + M * D, Sc);

  // 4) softmax rows -> packed f16 P
  attn_softmax_rows<<<(unsigned)(B * S), 256, 0, stream>>>(Sc);

  // 5) context = P V per batch, f32 out
  attn_context_gemm<<<dim3((unsigned)(D / 128), (unsigned)(S / 128), (unsigned)B), 256, 0, stream>>>(
      (const _Float16*)Sc, QKV + 2 * M * D, (float*)d_out);
}
